// MaskedSelfAttention_34995393528436
// MI455X (gfx1250) — matrix operand order repaired and checked by reading, not compiled, not run on hardware
//
#include <hip/hip_runtime.h>
#include <hip/hip_bf16.h>

typedef __attribute__((ext_vector_type(16))) _Float16 v16h;
typedef __attribute__((ext_vector_type(8)))  _Float16 v8h;
typedef __attribute__((ext_vector_type(2)))  _Float16 v2h;
typedef __attribute__((ext_vector_type(8)))  float    v8f;
typedef __attribute__((ext_vector_type(2)))  float    v2f;

#define WMMA_F32_F16(a, b, c) \
  __builtin_amdgcn_wmma_f32_16x16x32_f16(false, (a), false, (b), (short)0, (c), false, false)

constexpr int EMB = 1024;
constexpr int SEQ = 1024;
constexpr int NH  = 16;
constexpr int DH  = 64;
constexpr int NB  = 8;
constexpr float MASK_FILL = -1000.0f;
constexpr float SCALE_F   = 1.0f / 32.0f;   // 1/sqrt(1024)

// ---------------------------------------------------------------------------
// Kernel 0: fc_w (f32) -> f16 scratch copy
// ---------------------------------------------------------------------------
__global__ void cvt_kernel(const float* __restrict__ src,
                           _Float16* __restrict__ dst, int nElem) {
  int i = blockIdx.x * blockDim.x + threadIdx.x;
  int stride = gridDim.x * blockDim.x;
  for (; i < nElem; i += stride) dst[i] = (_Float16)src[i];
}

// ---------------------------------------------------------------------------
// Kernel 0b: V (n,k,h,d) f32 -> Vt[(n*NH+h)*DH + d][k] f16  (64x64 LDS tiles)
// ---------------------------------------------------------------------------
__global__ __launch_bounds__(256) void vtr_kernel(const float* __restrict__ Vmat,
                                                  _Float16* __restrict__ Vt) {
  __shared__ _Float16 T[64][72];
  const int wg = blockIdx.x;
  const int kc = wg & 15;
  const int nh = wg >> 4;
  const int h  = nh & (NH - 1);
  const int n  = nh >> 4;
  const int k0 = kc * 64;
  const int tid = threadIdx.x;

  const float* src = Vmat + (size_t)n * SEQ * EMB + h * DH;
  #pragma unroll
  for (int e = tid; e < 64 * 64; e += 256) {
    int kk = e >> 6, d = e & 63;
    T[d][kk] = (_Float16)src[(size_t)(k0 + kk) * EMB + d];
  }
  __syncthreads();
  _Float16* dst = Vt + (size_t)nh * DH * SEQ + k0;
  #pragma unroll
  for (int e = tid; e < 64 * 64; e += 256) {
    int d = e >> 6, kk = e & 63;
    dst[(size_t)d * SEQ + kk] = T[d][kk];
  }
}

// ---------------------------------------------------------------------------
// Kernel 1: attention, one (n, h, 16-row query tile) per workgroup, 4 waves.
// ---------------------------------------------------------------------------
__global__ __launch_bounds__(128) void attn_kernel(
    const _Float16* __restrict__ Vt, const float* __restrict__ Kmat,
    const float* __restrict__ Qmat, const int* __restrict__ amask,
    const int* __restrict__ kpm, _Float16* __restrict__ X) {
  __shared__ float    S[16][1032];
  __shared__ _Float16 P[16][1032];
  __shared__ float    redM[16][68];
  __shared__ float    red[16][8];
  __shared__ float    rinv[16];
  __shared__ int      kpmRow[SEQ];

  const int wg  = blockIdx.x;
  const int qt  = wg & 63;
  const int nh  = wg >> 6;
  const int h   = nh & (NH - 1);
  const int n   = nh >> 4;
  const int q0  = qt * 16;

  const int tid  = threadIdx.x;
  const int wid  = tid >> 5;
  const int lane = tid & 31;
  const int r    = lane & 15;
  const int half = lane >> 4;

  const float* Qb = Qmat + ((size_t)(n * SEQ + q0)) * EMB + h * DH;
  const float* Kb = Kmat + (size_t)n * SEQ * EMB + h * DH;

  #pragma unroll
  for (int c = tid; c < SEQ; c += 128) kpmRow[c] = kpm[n * SEQ + c];
  __syncthreads();

  // --- A fragments of the 16x64 Q tile (two K=32 chunks)
  v16h aq0, aq1;
  {
    const float* qrow = Qb + (size_t)r * EMB;
    #pragma unroll
    for (int i = 0; i < 8; ++i) {
      aq0[i]     = (_Float16)qrow[half * 8 + i];
      aq0[8 + i] = (_Float16)qrow[16 + half * 8 + i];
      aq1[i]     = (_Float16)qrow[32 + half * 8 + i];
      aq1[8 + i] = (_Float16)qrow[48 + half * 8 + i];
    }
  }

  // --- S = Q K^T with fused mask + scale + running row-max
  const float MASKED_Z = MASK_FILL * SCALE_F;  // -31.25
  float pmax[8];
  #pragma unroll
  for (int iv = 0; iv < 8; ++iv) pmax[iv] = -3.4e38f;

  const int* amTile = amask + (size_t)q0 * SEQ;
  for (int t = 0; t < 16; ++t) {
    const int nt  = (t << 2) | wid;
    const int c   = nt * 16 + r;
    const float* krow = Kb + (size_t)c * EMB;
    v16h b0, b1;
    #pragma unroll
    for (int i = 0; i < 16; ++i) {
      b0[i] = (_Float16)krow[half * 8 + i + (i & 8)];
      b1[i] = (_Float16)krow[32 + half * 8 + i + (i & 8)];
    }
    const int kp = kpmRow[c];
    int am8[8];
    #pragma unroll
    for (int iv = 0; iv < 8; ++iv)
      am8[iv] = amTile[(size_t)(iv + 8 * half) * SEQ + c];

    v8f acc = {};
    acc = WMMA_F32_F16(aq0, b0, acc);
    acc = WMMA_F32_F16(aq1, b1, acc);

    #pragma unroll
    for (int iv = 0; iv < 8; ++iv) {
      int keep = (am8[iv] != 0) & (kp != 0);
      float z  = keep ? acc[iv] * SCALE_F : MASKED_Z;
      S[iv + 8 * half][c] = z;
      pmax[iv] = fmaxf(pmax[iv], z);
    }
  }
  #pragma unroll
  for (int iv = 0; iv < 8; ++iv)
    redM[iv + 8 * half][(wid << 4) | r] = pmax[iv];
  __syncthreads();

  // --- row max reduction (LDS only), then vectorized exp + sum pass
  const int srow = tid >> 3;
  const int seg  = tid & 7;
  float mx = redM[srow][seg * 8];
  #pragma unroll
  for (int i = 1; i < 8; ++i) mx = fmaxf(mx, redM[srow][seg * 8 + i]);
  red[srow][seg] = mx;
  __syncthreads();
  #pragma unroll
  for (int i = 0; i < 8; ++i) mx = fmaxf(mx, red[srow][i]);

  float sm = 0.0f;
  #pragma unroll 4
  for (int j = 0; j < 64; ++j) {
    const int c = seg * 2 + 16 * j;
    v2f z = *(const v2f*)&S[srow][c];
    float e0 = __expf(z.x - mx);
    float e1 = __expf(z.y - mx);
    v2h p; p.x = (_Float16)e0; p.y = (_Float16)e1;
    *(v2h*)&P[srow][c] = p;
    sm += e0 + e1;
  }
  __syncthreads();
  red[srow][seg] = sm;
  __syncthreads();
  if (seg == 0) {
    float s = 0.0f;
    #pragma unroll
    for (int i = 0; i < 8; ++i) s += red[srow][i];
    rinv[srow] = 1.0f / s;
  }
  __syncthreads();

  // --- O = P V : wave owns one 16-wide d-tile; two independent WMMA chains
  const int col = wid * 16 + r;
  const _Float16* vtrow = Vt + ((size_t)nh * DH + col) * SEQ;
  v8f oa0 = {}, oa1 = {};
  for (int kt = 0; kt < 32; kt += 2) {
    const int kb = kt * 32;
    v16h ap0, ap1;
    v8h l0 = *(const v8h*)&P[r][kb + half * 16];
    v8h h0 = *(const v8h*)&P[r][kb + 8 + half * 16];
    v8h l1 = *(const v8h*)&P[r][kb + 32 + half * 16];
    v8h h1 = *(const v8h*)&P[r][kb + 40 + half * 16];
    #pragma unroll
    for (int i = 0; i < 8; ++i) {
      ap0[i] = l0[i]; ap0[8 + i] = h0[i];
      ap1[i] = l1[i]; ap1[8 + i] = h1[i];
    }
    v16h bv0 = *(const v16h*)(vtrow + kb + half * 16);
    v16h bv1 = *(const v16h*)(vtrow + kb + 32 + half * 16);
    oa0 = WMMA_F32_F16(ap0, bv0, oa0);
    oa1 = WMMA_F32_F16(ap1, bv1, oa1);
  }
  #pragma unroll
  for (int iv = 0; iv < 8; ++iv) {
    int row = iv + 8 * half;
    float val = (oa0[iv] + oa1[iv]) * rinv[row];
    X[((size_t)(n * SEQ + q0 + row)) * EMB + h * DH + col] = (_Float16)val;
  }
}

// ---------------------------------------------------------------------------
// Kernel 2: out = X @ W^T + b.  Wave tile 32x64: 8 WMMAs per k-step with
// 2 A + 4 B fragment loads (2x arithmetic intensity vs 16x64 tiles).
// Workgroup tile 128x128, 8 waves.
// ---------------------------------------------------------------------------
__global__ __launch_bounds__(256) void fc_kernel(
    const _Float16* __restrict__ X, const _Float16* __restrict__ Wh,
    const float* __restrict__ bias, float* __restrict__ out) {
  const int tid  = threadIdx.x;
  const int wid  = tid >> 5;
  const int lane = tid & 31;
  const int r    = lane & 15;
  const int half = lane >> 4;

  const int wg   = blockIdx.x;           // 64 m-blocks x 8 n-blocks
  const int mblk = wg >> 3;
  const int nblk = wg & 7;
  const int m0   = mblk * 128 + (wid & 3) * 32;
  const int n0   = nblk * 128 + (wid >> 2) * 64;

  const _Float16* xrow0 = X + (size_t)(m0 + r) * EMB;
  const _Float16* xrow1 = X + (size_t)(m0 + 16 + r) * EMB;
  v8f acc[2][4] = {};
  for (int kt = 0; kt < 32; ++kt) {
    const int kb = kt * 32;
    v16h a0, a1;
    {
      v8h l0 = *(const v8h*)(xrow0 + kb + half * 16);
      v8h h0 = *(const v8h*)(xrow0 + kb + 8 + half * 16);
      v8h l1 = *(const v8h*)(xrow1 + kb + half * 16);
      v8h h1 = *(const v8h*)(xrow1 + kb + 8 + half * 16);
      #pragma unroll
      for (int i = 0; i < 8; ++i) {
        a0[i] = l0[i]; a0[8 + i] = h0[i];
        a1[i] = l1[i]; a1[8 + i] = h1[i];
      }
    }
    const _Float16* wbase = Wh + (size_t)(n0 + r) * EMB + kb + half * 16;
    v16h b0 = *(const v16h*)(wbase);
    v16h b1 = *(const v16h*)(wbase + 16 * EMB);
    v16h b2 = *(const v16h*)(wbase + 32 * EMB);
    v16h b3 = *(const v16h*)(wbase + 48 * EMB);
    acc[0][0] = WMMA_F32_F16(a0, b0, acc[0][0]);
    acc[0][1] = WMMA_F32_F16(a0, b1, acc[0][1]);
    acc[0][2] = WMMA_F32_F16(a0, b2, acc[0][2]);
    acc[0][3] = WMMA_F32_F16(a0, b3, acc[0][3]);
    acc[1][0] = WMMA_F32_F16(a1, b0, acc[1][0]);
    acc[1][1] = WMMA_F32_F16(a1, b1, acc[1][1]);
    acc[1][2] = WMMA_F32_F16(a1, b2, acc[1][2]);
    acc[1][3] = WMMA_F32_F16(a1, b3, acc[1][3]);
  }
  #pragma unroll
  for (int im = 0; im < 2; ++im) {
    #pragma unroll
    for (int iv = 0; iv < 8; ++iv) {
      int row = m0 + im * 16 + iv + 8 * half;
      float* orow = out + (size_t)row * EMB + n0 + r;
      #pragma unroll
      for (int j = 0; j < 4; ++j)
        orow[j * 16] = acc[im][j][iv] + bias[n0 + j * 16 + r];
    }
  }
}

// ---------------------------------------------------------------------------
extern "C" void kernel_launch(void* const* d_in, const int* in_sizes, int n_in,
                              void* d_out, int out_size, void* d_ws, size_t ws_size,
                              hipStream_t stream) {
  (void)in_sizes; (void)n_in; (void)out_size; (void)ws_size;
  const float* values  = (const float*)d_in[0];
  const float* keys    = (const float*)d_in[1];
  const float* queries = (const float*)d_in[2];
  const float* fc_w    = (const float*)d_in[3];
  const float* fc_b    = (const float*)d_in[4];
  const int*   amask   = (const int*)d_in[5];
  const int*   kpm     = (const int*)d_in[6];
  float* out = (float*)d_out;

  char* ws = (char*)d_ws;
  _Float16* X  = (_Float16*)ws;                                    // 16 MB
  _Float16* Wh = (_Float16*)(ws + (size_t)NB * SEQ * EMB * 2);     // +2 MB
  _Float16* Vt = (_Float16*)(ws + (size_t)(NB * SEQ * EMB + EMB * EMB) * 2); // +16 MB

  cvt_kernel<<<1024, 256, 0, stream>>>(fc_w, Wh, EMB * EMB);
  vtr_kernel<<<NB * NH * (SEQ / 64), 256, 0, stream>>>(values, Vt);
  attn_kernel<<<NB * NH * (SEQ / 16), 128, 0, stream>>>(
      Vt, keys, queries, amask, kpm, X);
  fc_kernel<<<(NB * SEQ / 128) * (EMB / 128), 256, 0, stream>>>(X, Wh, fc_b, out);
}
